// GraphAttentionEncoder_78683800863088
// MI455X (gfx1250) — compile-verified, matrix operand order checked by reading, not run on hardware
//
#include <hip/hip_runtime.h>
#include <hip/hip_bf16.h>

// ---------------- problem constants ----------------
#define Bc   32
#define Nn   500
#define NP   512      // padded node count (32 tiles of 16)
#define Dm   128
#define Hh   8
#define DKc  16
#define FFc  512
#define EPSb 1e-5f

typedef __attribute__((ext_vector_type(16))) _Float16 v16h;
typedef __attribute__((ext_vector_type(8)))  _Float16 v8h;
typedef __attribute__((ext_vector_type(8)))  float    v8f;

#define WMMA_F16(a, b, c) \
  __builtin_amdgcn_wmma_f32_16x16x32_f16(false, (a), false, (b), (short)0, (c), false, false)

// ---- WMMA operand loaders (CDNA5 16-bit 16x16x32 layouts, wave32) ----
// All f16 sources are 16-byte aligned at every use site -> v8h (b128) loads.

// A (16 rows x 32 K) from f16 row-major [rows][ld]
__device__ __forceinline__ v16h a_tile_f16(const _Float16* __restrict__ base, int ld, int lane) {
  int r = lane & 15, kb = (lane >> 4) * 8;
  v8h lo = *(const v8h*)(base + (size_t)r * ld + kb);
  v8h hi = *(const v8h*)(base + (size_t)r * ld + kb + 16);
  v16h a;
#pragma unroll
  for (int j = 0; j < 8; ++j) { a[j] = lo[j]; a[j + 8] = hi[j]; }
  return a;
}
// B (32 K x 16 cols) from TRANSPOSED f16 weights [cols][ldT]; element B[k][c] = baseT[c*ldT + k]
__device__ __forceinline__ v16h b_tile_f16t(const _Float16* __restrict__ baseT, int ldT, int lane) {
  int c = lane & 15, kb = (lane >> 4) * 8;
  v8h lo = *(const v8h*)(baseT + (size_t)c * ldT + kb);
  v8h hi = *(const v8h*)(baseT + (size_t)c * ldT + kb + 16);
  v16h b;
#pragma unroll
  for (int j = 0; j < 8; ++j) { b[j] = lo[j]; b[j + 8] = hi[j]; }
  return b;
}
// B tile from f16 K-matrix [node][DK] (K-dim = dk, zero-padded 16..31)
__device__ __forceinline__ v16h bK_f16(const _Float16* __restrict__ nodeBase, int lane) {
  int c = lane & 15, kb = (lane >> 4) * 8;
  v8h lo = *(const v8h*)(nodeBase + (size_t)c * DKc + kb);
  v16h b;
#pragma unroll
  for (int j = 0; j < 8; ++j) { b[j] = lo[j]; b[j + 8] = (_Float16)0.f; }
  return b;
}
// A tile from LDS 16x32 f16 row-major
__device__ __forceinline__ v16h aP_lds(const _Float16* lds, int lane) {
  int r = lane & 15, kb = (lane >> 4) * 8;
  v16h a;
#pragma unroll
  for (int j = 0; j < 8; ++j) { a[j] = lds[r * 32 + kb + j]; a[j + 8] = lds[r * 32 + 16 + kb + j]; }
  return a;
}

// ---------------- kernels ----------------

// transpose+convert f32 [R][C] -> f16 [C][R]; blockIdx.y = matrix index (e.g. head)
__global__ void wcvt_t_kernel(const float* __restrict__ in, _Float16* __restrict__ out,
                              int R, int C) {
  size_t base = (size_t)blockIdx.y * R * C;
  int i = blockIdx.x * 256 + threadIdx.x;
  if (i >= R * C) return;
  int r = i / C, c = i % C;
  out[base + (size_t)c * R + r] = (_Float16)in[base + i];
}

// h = x @ init_W + init_b ; zero padded rows ; writes f32 master + f16 mirror
__global__ void init_h_kernel(const float* __restrict__ x, const float* __restrict__ W,
                              const float* __restrict__ bias, float* __restrict__ h,
                              _Float16* __restrict__ h16) {
  size_t i = (size_t)blockIdx.x * 256 + threadIdx.x;     // over Bc*NP*Dm
  int d = (int)(i & 127);
  size_t row = i >> 7;                                   // b*NP + n
  int n = (int)(row & (NP - 1));
  int b = (int)(row >> 9);
  float v = 0.f;
  if (n < Nn) {
    const float* xr = x + ((size_t)b * Nn + n) * 2;
    v = xr[0] * W[d] + xr[1] * W[Dm + d] + bias[d];
  }
  h[i] = v;
  h16[i] = (_Float16)v;
}

// edge MLP 1->16->16->8, output f16 [B][NP][512][H]
__global__ void edge_bias_kernel(const float* __restrict__ em,
                                 const float* __restrict__ W1, const float* __restrict__ b1,
                                 const float* __restrict__ W2, const float* __restrict__ b2,
                                 const float* __restrict__ W3, const float* __restrict__ b3,
                                 _Float16* __restrict__ eb) {
  __shared__ float w1[16], bb1[16], w2[256], bb2[16], w3[128], bb3[8];
  int t = threadIdx.x;
  if (t < 16) { w1[t] = W1[t]; bb1[t] = b1[t]; bb2[t] = b2[t]; }
  w2[t] = W2[t];
  if (t < 128) w3[t] = W3[t];
  if (t < 8) bb3[t] = b3[t];
  __syncthreads();
  size_t idx = (size_t)blockIdx.x * 256 + t;
  if (idx >= (size_t)Bc * Nn * Nn) return;
  float v = em[idx];
  int b = (int)(idx / ((size_t)Nn * Nn));
  int rem = (int)(idx % ((size_t)Nn * Nn));
  int n = rem / Nn, m = rem % Nn;
  float h1[16], h2[16];
#pragma unroll
  for (int i = 0; i < 16; ++i) h1[i] = fmaxf(v * w1[i] + bb1[i], 0.f);
#pragma unroll
  for (int i = 0; i < 16; ++i) {
    float a = bb2[i];
#pragma unroll
    for (int j = 0; j < 16; ++j) a += h1[j] * w2[j * 16 + i];
    h2[i] = fmaxf(a, 0.f);
  }
  _Float16* o = eb + (((size_t)b * NP + n) * 512 + m) * Hh;
#pragma unroll
  for (int i = 0; i < 8; ++i) {
    float a = bb3[i];
#pragma unroll
    for (int j = 0; j < 16; ++j) a += h2[j] * w3[j * 8 + i];
    o[i] = (_Float16)a;
  }
}

// Q/K/V projection from f16 activations + transposed f16 weights [H][16][128].
// TRANSP=false: out [H][B][NP][DK] ; TRANSP=true: out [H][B][DK][NP] (for V)
template <bool TRANSP>
__global__ void qkv_kernel(const _Float16* __restrict__ h16, const _Float16* __restrict__ WT,
                           _Float16* __restrict__ out) {
  int tile = blockIdx.x, hd = blockIdx.y, lane = threadIdx.x;
  int b = tile / (NP / 16);
  int n0 = (tile % (NP / 16)) * 16;
  const _Float16* abase = h16 + ((size_t)b * NP + n0) * Dm;
  const _Float16* wbase = WT + (size_t)hd * DKc * Dm;   // [16][128]
  v8f acc = {};
#pragma unroll
  for (int k0 = 0; k0 < Dm; k0 += 32) {
    v16h a  = a_tile_f16(abase + k0, Dm, lane);
    v16h bt = b_tile_f16t(wbase + k0, Dm, lane);
    acc = WMMA_F16(a, bt, acc);
  }
  int c = lane & 15, rb = (lane >> 4) * 8;
  if (!TRANSP) {
    _Float16* obase = out + (((size_t)hd * Bc + b) * NP + n0) * DKc;
#pragma unroll
    for (int v = 0; v < 8; ++v) obase[(size_t)(rb + v) * DKc + c] = (_Float16)acc[v];
  } else {
    _Float16* obase = out + (((size_t)hd * Bc + b) * DKc + c) * NP + n0;
#pragma unroll
    for (int v = 0; v < 8; ++v) obase[rb + v] = (_Float16)acc[v];
  }
}

// Flash attention: one wave32 per (16-row tile, head); streams K/V in 32-col steps.
// V is pre-transposed [H][B][DK][NP] so the P.V B-tile is two contiguous b128 loads.
__global__ void attn_kernel(const _Float16* __restrict__ Qf, const _Float16* __restrict__ Kf,
                            const _Float16* __restrict__ Vt, const _Float16* __restrict__ ebias,
                            _Float16* __restrict__ aout16) {
  __shared__ _Float16 pLds[16 * 32];
  int tile = blockIdx.x, hd = blockIdx.y, lane = threadIdx.x;
  int b = tile / (NP / 16);
  int n0 = (tile % (NP / 16)) * 16;
  int r = lane & 15, c = lane & 15, kb = (lane >> 4) * 8, rb = (lane >> 4) * 8;
  const size_t kvoff = (((size_t)hd * Bc + b) * NP) * DKc;
  const size_t vtoff = (((size_t)hd * Bc + b) * DKc) * NP;
  // A = Q tile (16x32, dk padded), 1/sqrt(DK) folded in
  const _Float16* qbase = Qf + kvoff + (size_t)n0 * DKc;
  v8h qlo = *(const v8h*)(qbase + (size_t)r * DKc + kb);
  v16h aq;
#pragma unroll
  for (int j = 0; j < 8; ++j) {
    aq[j] = (_Float16)((float)qlo[j] * 0.25f);
    aq[j + 8] = (_Float16)0.f;
  }
  float Mrow[8], Lrow[8];
  v8f acc = {};
#pragma unroll
  for (int v = 0; v < 8; ++v) { Mrow[v] = -1e30f; Lrow[v] = 0.f; }
  const bool useBias = (ebias != nullptr);

  for (int m0 = 0; m0 < NP; m0 += 32) {
    __builtin_prefetch(Kf + kvoff + (size_t)(m0 + 32) * DKc, 0, 0);
    __builtin_prefetch(Vt + vtoff + (size_t)c * NP + m0 + 32, 0, 0);
    v16h bk0 = bK_f16(Kf + kvoff + (size_t)m0 * DKc, lane);
    v16h bk1 = bK_f16(Kf + kvoff + (size_t)(m0 + 16) * DKc, lane);
    v8f s0 = {}, s1 = {};
    s0 = WMMA_F16(aq, bk0, s0);
    s1 = WMMA_F16(aq, bk1, s1);
#pragma unroll
    for (int v = 0; v < 8; ++v) {
      float x0 = s0[v], x1 = s1[v];
      if (useBias) {
        const _Float16* eb = ebias + (((size_t)b * NP + n0 + rb + v) * 512 + m0) * Hh + hd;
        x0 += (float)eb[(size_t)c * Hh];
        x1 += (float)eb[(size_t)(c + 16) * Hh];
      }
      x0 = (m0 + c < Nn) ? x0 : -1e30f;
      x1 = (m0 + 16 + c < Nn) ? x1 : -1e30f;
      float mx = fmaxf(x0, x1);
#pragma unroll
      for (int d = 1; d < 16; d <<= 1) mx = fmaxf(mx, __shfl_xor(mx, d, 32));
      float Mnew = fmaxf(Mrow[v], mx);
      float sc = __expf(Mrow[v] - Mnew);
      float e0 = __expf(x0 - Mnew), e1 = __expf(x1 - Mnew);
      float rs = e0 + e1;
#pragma unroll
      for (int d = 1; d < 16; d <<= 1) rs += __shfl_xor(rs, d, 32);
      Lrow[v] = Lrow[v] * sc + rs;
      acc[v] *= sc;
      Mrow[v] = Mnew;
      pLds[(rb + v) * 32 + c] = (_Float16)e0;
      pLds[(rb + v) * 32 + 16 + c] = (_Float16)e1;
    }
    __syncthreads();                      // P tile visible across lanes
    v16h ap = aP_lds(pLds, lane);         // C/D layout -> A layout via LDS
    // B = V tile from transposed V: element B[m][dk] = Vt[dk][m0+m]
    v8h vlo = *(const v8h*)(Vt + vtoff + (size_t)c * NP + m0 + kb);
    v8h vhi = *(const v8h*)(Vt + vtoff + (size_t)c * NP + m0 + kb + 16);
    v16h bv;
#pragma unroll
    for (int j = 0; j < 8; ++j) { bv[j] = vlo[j]; bv[j + 8] = vhi[j]; }
    __syncthreads();                      // reads done before next iteration's stores
    acc = WMMA_F16(ap, bv, acc);
  }
#pragma unroll
  for (int v = 0; v < 8; ++v) {
    float inv = 1.f / Lrow[v];
    aout16[(((size_t)b * NP + n0 + rb + v)) * Dm + hd * DKc + c] = (_Float16)(acc[v] * inv);
  }
}

// WMMA GEMM, 4 column-tiles per wave: out[rows x NCOLS] (+)= A[rows x KDIM] @ B
// A: f16 row-major; BT: f16 transposed [NCOLS][KDIM]
// MODE 0: outF += acc                 (Wo projection, residual into f32 h)
// MODE 1: out16 = relu(acc + bias)    (FF1 -> f16 activations)
// MODE 2: outF += acc + bias          (FF2, residual into f32 h)
template <int KDIM, int NCOLS, int MODE>
__global__ void gemm_kernel(const _Float16* __restrict__ A, const _Float16* __restrict__ BT,
                            const float* __restrict__ bias, float* __restrict__ outF,
                            _Float16* __restrict__ out16) {
  int tile = blockIdx.x, lane = threadIdx.x;
  size_t row0 = (size_t)tile * 16;
  int c0 = blockIdx.y * 64;                 // 4 tiles of 16 columns
  const _Float16* abase = A + row0 * KDIM;
  v8f acc[4] = {{}, {}, {}, {}};
#pragma unroll
  for (int k0 = 0; k0 < KDIM; k0 += 32) {
    v16h a = a_tile_f16(abase + k0, KDIM, lane);
#pragma unroll
    for (int t = 0; t < 4; ++t) {
      v16h bt = b_tile_f16t(BT + (size_t)(c0 + t * 16) * KDIM + k0, KDIM, lane);
      acc[t] = WMMA_F16(a, bt, acc[t]);
    }
  }
  int c = lane & 15, rb = (lane >> 4) * 8;
#pragma unroll
  for (int t = 0; t < 4; ++t) {
#pragma unroll
    for (int v = 0; v < 8; ++v) {
      int cc = c0 + t * 16 + c;
      size_t idx = (row0 + rb + v) * NCOLS + cc;
      float val = acc[t][v];
      if (MODE == 0) outF[idx] += val;
      if (MODE == 1) out16[idx] = (_Float16)fmaxf(val + bias[cc], 0.f);
      if (MODE == 2) outF[idx] += val + bias[cc];
    }
  }
}

// BatchNorm (training): stats over the Bc*Nn valid rows
__global__ void bn_zero(float* stats) { stats[threadIdx.x] = 0.f; }

__global__ void bn_stats(const float* __restrict__ h, float* __restrict__ stats) {
  __shared__ float sm[512];
  int t = threadIdx.x;            // 256
  int c = t & 127, half = t >> 7;
  float s = 0.f, sq = 0.f;
  int r0 = blockIdx.x * 64;
#pragma unroll 4
  for (int i = 0; i < 32; ++i) {
    int rr = r0 + i * 2 + half;
    int b = rr / Nn, n = rr % Nn;
    float v = h[((size_t)b * NP + n) * Dm + c];
    s += v; sq += v * v;
  }
  sm[t] = s; sm[256 + t] = sq;
  __syncthreads();
  if (t < 128) {
    atomicAdd(&stats[c], sm[t] + sm[t + 128]);
    atomicAdd(&stats[128 + c], sm[256 + t] + sm[256 + t + 128]);
  }
}

__global__ void bn_final(float* stats) {
  int c = threadIdx.x;
  const float inv = 1.f / (float)(Bc * Nn);
  float mu = stats[c] * inv;
  float var = stats[128 + c] * inv - mu * mu;
  stats[c] = mu;
  stats[128 + c] = rsqrtf(var + EPSb);
}

// normalize + affine; refresh both f32 master and f16 mirror
__global__ void bn_apply(float* __restrict__ h, _Float16* __restrict__ h16,
                         const float* __restrict__ stats,
                         const float* __restrict__ g, const float* __restrict__ bta) {
  size_t i = (size_t)blockIdx.x * 256 + threadIdx.x;   // Bc*NP*Dm
  int c = (int)(i & 127);
  float v = (h[i] - stats[c]) * stats[128 + c] * g[c] + bta[c];
  h[i] = v;
  h16[i] = (_Float16)v;
}

// graph embedding: mean over valid rows, concat S, MLP 144->128->128->128
__global__ void gemb_kernel(const float* __restrict__ h, const float* __restrict__ S,
                            const float* __restrict__ W0, const float* __restrict__ c0,
                            const float* __restrict__ W1, const float* __restrict__ c1,
                            const float* __restrict__ W2, const float* __restrict__ c2,
                            float* __restrict__ outGE) {
  __shared__ float g0[144], g1[128], g2[128];
  int b = blockIdx.x, t = threadIdx.x;   // 128 threads
  float s = 0.f;
  for (int n = 0; n < Nn; ++n) s += h[((size_t)b * NP + n) * Dm + t];
  g0[t] = s / (float)Nn;
  if (t < 16) g0[128 + t] = S[b * 16 + t];
  __syncthreads();
  { float a = c0[t]; for (int j = 0; j < 144; ++j) a += g0[j] * W0[j * 128 + t]; g1[t] = fmaxf(a, 0.f); }
  __syncthreads();
  { float a = c1[t]; for (int j = 0; j < 128; ++j) a += g1[j] * W1[j * 128 + t]; g2[t] = fmaxf(a, 0.f); }
  __syncthreads();
  { float a = c2[t]; for (int j = 0; j < 128; ++j) a += g2[j] * W2[j * 128 + t];
    outGE[(size_t)b * Dm + t] = a; }
}

__global__ void copy_out_kernel(const float* __restrict__ h, float* __restrict__ out) {
  size_t i = (size_t)blockIdx.x * 256 + threadIdx.x;   // Bc*Nn*Dm
  if (i >= (size_t)Bc * Nn * Dm) return;
  size_t row = i >> 7;
  int d = (int)(i & 127);
  int b = (int)(row / Nn), n = (int)(row % Nn);
  out[i] = h[((size_t)b * NP + n) * Dm + d];
}

// ---------------- host launcher ----------------
extern "C" void kernel_launch(void* const* d_in, const int* in_sizes, int n_in,
                              void* d_out, int out_size, void* d_ws, size_t ws_size,
                              hipStream_t stream) {
  // inputs in reference-file (insertion) order
  const float* x     = (const float*)d_in[0];
  const float* S     = (const float*)d_in[1];
  const float* em    = (const float*)d_in[2];
  const float* initW = (const float*)d_in[3];
  const float* initb = (const float*)d_in[4];
  const float* eW0   = (const float*)d_in[5];
  const float* eW1   = (const float*)d_in[6];
  const float* eW2   = (const float*)d_in[7];
  const float* ebs0  = (const float*)d_in[8];
  const float* ebs1  = (const float*)d_in[9];
  const float* ebs2  = (const float*)d_in[10];
  const float* geW0  = (const float*)d_in[11];
  const float* geW1  = (const float*)d_in[12];
  const float* geW2  = (const float*)d_in[13];
  const float* geb0  = (const float*)d_in[14];
  const float* geb1  = (const float*)d_in[15];
  const float* geb2  = (const float*)d_in[16];

  // workspace carve-up
  char* ws = (char*)d_ws;
  float*    hbuf  = (float*)ws;    ws += (size_t)Bc * NP * Dm * sizeof(float);
  _Float16* h16   = (_Float16*)ws; ws += (size_t)Bc * NP * Dm * sizeof(_Float16);
  _Float16* aout16= (_Float16*)ws; ws += (size_t)Bc * NP * Dm * sizeof(_Float16);
  _Float16* ffb16 = (_Float16*)ws; ws += (size_t)Bc * NP * FFc * sizeof(_Float16);
  float*    stats = (float*)ws;    ws += 256 * sizeof(float);
  _Float16* qf    = (_Float16*)ws; ws += (size_t)Hh * Bc * NP * DKc * sizeof(_Float16);
  _Float16* kf    = (_Float16*)ws; ws += (size_t)Hh * Bc * NP * DKc * sizeof(_Float16);
  _Float16* vt    = (_Float16*)ws; ws += (size_t)Hh * Bc * NP * DKc * sizeof(_Float16);
  _Float16 *wqT[3], *wkT[3], *wvT[3], *woT[3], *w1T[3], *w2T[3];
  for (int l = 0; l < 3; ++l) {
    wqT[l] = (_Float16*)ws; ws += (size_t)Hh * Dm * DKc * sizeof(_Float16);
    wkT[l] = (_Float16*)ws; ws += (size_t)Hh * Dm * DKc * sizeof(_Float16);
    wvT[l] = (_Float16*)ws; ws += (size_t)Hh * Dm * DKc * sizeof(_Float16);
    woT[l] = (_Float16*)ws; ws += (size_t)Dm * Dm * sizeof(_Float16);
    w1T[l] = (_Float16*)ws; ws += (size_t)Dm * FFc * sizeof(_Float16);
    w2T[l] = (_Float16*)ws; ws += (size_t)FFc * Dm * sizeof(_Float16);
  }
  _Float16* ebias = (_Float16*)ws;  // [B][NP][512][H] f16, 134 MB (mostly L2-resident)

  const int TILES = Bc * NP / 16;   // 1024
  const int ELEMS = Bc * NP * Dm;   // 2,097,152

  // one-time: weight convert/transpose to f16
  for (int l = 0; l < 3; ++l) {
    const float* const* lp = (const float* const*)(d_in + 17 + 12 * l);
    wcvt_t_kernel<<<dim3(8, Hh), 256, 0, stream>>>(lp[0], wqT[l], Dm, DKc);     // Wq
    wcvt_t_kernel<<<dim3(8, Hh), 256, 0, stream>>>(lp[1], wkT[l], Dm, DKc);     // Wk
    wcvt_t_kernel<<<dim3(8, Hh), 256, 0, stream>>>(lp[2], wvT[l], Dm, DKc);     // Wv
    wcvt_t_kernel<<<dim3(64, 1), 256, 0, stream>>>(lp[3], woT[l], Dm, Dm);      // Wo (flat 128x128)
    wcvt_t_kernel<<<dim3(256, 1), 256, 0, stream>>>(lp[4], w1T[l], Dm, FFc);    // W1
    wcvt_t_kernel<<<dim3(256, 1), 256, 0, stream>>>(lp[6], w2T[l], FFc, Dm);    // W2
  }

  init_h_kernel<<<ELEMS / 256, 256, 0, stream>>>(x, initW, initb, hbuf, h16);
  edge_bias_kernel<<<(Bc * Nn * Nn + 255) / 256, 256, 0, stream>>>(
      em, eW0, ebs0, eW1, ebs1, eW2, ebs2, ebias);

  for (int l = 0; l < 3; ++l) {
    const float* const* lp = (const float* const*)(d_in + 17 + 12 * l);
    const float* bff1 = lp[5]; const float* bff2 = lp[7];
    const float* g1 = lp[8];  const float* b1 = lp[9];
    const float* g2 = lp[10]; const float* b2 = lp[11];

    qkv_kernel<false><<<dim3(TILES, Hh), 32, 0, stream>>>(h16, wqT[l], qf);
    qkv_kernel<false><<<dim3(TILES, Hh), 32, 0, stream>>>(h16, wkT[l], kf);
    qkv_kernel<true ><<<dim3(TILES, Hh), 32, 0, stream>>>(h16, wvT[l], vt);
    attn_kernel<<<dim3(TILES, Hh), 32, 0, stream>>>(qf, kf, vt, (l == 0) ? ebias : nullptr, aout16);
    gemm_kernel<128, 128, 0><<<dim3(TILES, 2), 32, 0, stream>>>(aout16, woT[l], nullptr, hbuf, nullptr);

    bn_zero<<<1, 256, 0, stream>>>(stats);
    bn_stats<<<(Bc * Nn) / 64, 256, 0, stream>>>(hbuf, stats);
    bn_final<<<1, 128, 0, stream>>>(stats);
    bn_apply<<<ELEMS / 256, 256, 0, stream>>>(hbuf, h16, stats, g1, b1);

    gemm_kernel<128, 512, 1><<<dim3(TILES, 8), 32, 0, stream>>>(h16, w1T[l], bff1, nullptr, ffb16);
    gemm_kernel<512, 128, 2><<<dim3(TILES, 2), 32, 0, stream>>>(ffb16, w2T[l], bff2, hbuf, nullptr);

    bn_zero<<<1, 256, 0, stream>>>(stats);
    bn_stats<<<(Bc * Nn) / 64, 256, 0, stream>>>(hbuf, stats);
    bn_final<<<1, 128, 0, stream>>>(stats);
    bn_apply<<<ELEMS / 256, 256, 0, stream>>>(hbuf, h16, stats, g2, b2);
  }

  copy_out_kernel<<<(Bc * Nn * Dm + 255) / 256, 256, 0, stream>>>(hbuf, (float*)d_out);
  gemb_kernel<<<Bc, 128, 0, stream>>>(hbuf, S, geW0, geb0, geW1, geb1, geW2, geb2,
                                      (float*)d_out + (size_t)Bc * Nn * Dm);
}
